// LSTMDecoder_85899345931
// MI455X (gfx1250) — matrix operand order repaired and checked by reading, compiled, not run on hardware
//
#include <hip/hip_runtime.h>
#include <hip/hip_bf16.h>

// ---------------------------------------------------------------------------
// Persistent LSTM decoder for MI455X (gfx1250, wave32, bf16 WMMA, f32 accum)
// B=256, H=1024, F=512, T=128
// z_t = h_{t-1} @ (W+U) + b  (t>=1);  z_0 = [x0,h0] @ [W;U] + b
//
// Design:
//  - ONE persistent kernel runs all 128 recurrent steps (grid barrier between
//    steps) -> no per-step launch latency on the serial dependence chain.
//  - Each block owns a fixed 16-column slice of H for all 4 gates; its weight
//    slice (128 KB) is loaded ONCE into LDS via global_load_async_to_lds_b128
//    and reused for 127 steps -> weights never re-cross L2 after step 0.
//  - Cell state c lives entirely in registers (fixed tile ownership).
//  - Gates fused into the GEMM epilogue; h stored bf16 into seq for the next
//    step and the final emission GEMM.
// ---------------------------------------------------------------------------

#define B_   256
#define H_   1024
#define F_   512
#define T_   128
#define FH_  4096            // 4*H
#define SEQ_STRIDE (T_ * H_) // row stride of seq[B,T,H]
#define NBLOCKS 128u         // persistent grid: (2, 64)

typedef __attribute__((ext_vector_type(8)))  float   v8f;
typedef __attribute__((ext_vector_type(8)))  __bf16  v8bf;
typedef __attribute__((ext_vector_type(16))) __bf16  v16bf;
typedef __attribute__((ext_vector_type(4)))  int     v4i;

#if defined(__AMDGCN__) && \
    __has_builtin(__builtin_amdgcn_global_load_async_to_lds_b128) && \
    __has_builtin(__builtin_amdgcn_s_wait_asynccnt)
#define USE_ASYNC_LDS 1
// Address-space-qualified int4 pointer types matching the builtin signature.
typedef __attribute__((address_space(1))) v4i gv4i_t;
typedef __attribute__((address_space(3))) v4i lv4i_t;
#endif

static __device__ __forceinline__ __bf16 f2b(float f) {
    union { float f; unsigned u; } cv; cv.f = f;
    unsigned u = cv.u;
    unsigned r = (u + 0x7FFFu + ((u >> 16) & 1u)) >> 16;   // round-to-nearest-even
    union { unsigned short s; __bf16 b; } cb; cb.s = (unsigned short)r;
    return cb.b;
}

static __device__ __forceinline__ float sigm(float x) {
    return 1.0f / (1.0f + __expf(-x));
}
static __device__ __forceinline__ float tanh_fast(float x) {
    x = fminf(fmaxf(x, -15.0f), 15.0f);
    float e = __expf(-2.0f * x);
    return (1.0f - e) / (1.0f + e);
}

// B-matrix (KxN, 16-bit) WMMA fragment K index for (lane, slot).
static __device__ __forceinline__ int bK(int lane, int slot) {
    return ((lane < 16) ? 0 : 8) + ((slot < 8) ? slot : (slot + 8));
}

// A fragment (16x32 bf16 from row-major source): two contiguous 16B loads.
static __device__ __forceinline__ v16bf load_a_frag(const __bf16* rowp, int kbase, bool hi) {
    const __bf16* p0 = rowp + kbase + (hi ? 8 : 0);
    v8bf lo = *(const v8bf*)p0;
    v8bf hv = *(const v8bf*)(p0 + 16);
    v16bf r;
#pragma unroll
    for (int i = 0; i < 8; ++i) { r[i] = lo[i]; r[i + 8] = hv[i]; }
    return r;
}

// Pre-swizzled B fragment (global or LDS): lane reads 16 contiguous bf16.
static __device__ __forceinline__ v16bf load_b_frag(const __bf16* fragbase, int lane) {
    const __bf16* p = fragbase + lane * 16;
    v8bf lo = *(const v8bf*)p;
    v8bf hv = *(const v8bf*)(p + 8);
    v16bf r;
#pragma unroll
    for (int i = 0; i < 8; ++i) { r[i] = lo[i]; r[i + 8] = hv[i]; }
    return r;
}

// ---------------------------------------------------------------------------
// Prep: bf16 fragment-swizzled weights, bf16 [x0,h0], zero grid barrier.
//  Wcat frags: [g(4)][nt(64)][kt(64)][512]   (step 0, K=2048: W rows then U)
//  Wc   frags: [nt(64)][g(4)][kt(32)][512]   (W+U; nt-major so each block's
//                                             128 KB LDS slice is contiguous)
//  Wd   frags: [nt(32)][kt(32)][512]
// ---------------------------------------------------------------------------
__global__ void __launch_bounds__(256)
prep_kernel(const float* __restrict__ x0, const float* __restrict__ h0,
            const float* __restrict__ W,  const float* __restrict__ U,
            const float* __restrict__ Wd,
            __bf16* __restrict__ wcat, __bf16* __restrict__ wc,
            __bf16* __restrict__ wdf,  __bf16* __restrict__ xh,
            unsigned* __restrict__ bar)
{
    const long tid = (long)blockIdx.x * blockDim.x + threadIdx.x;
    const long nth = (long)gridDim.x * blockDim.x;

    // Wcat fragments (step 0, K=2048), layout [g][nt][kt64][512]
    for (long j = tid; j < 4L * 64 * 64 * 512; j += nth) {
        int e = (int)(j & 511); int lane = e >> 4; int slot = e & 15;
        long t = j >> 9; int kt = (int)(t & 63); t >>= 6; int nt = (int)(t & 63); int g = (int)(t >> 6);
        int k   = kt * 32 + bK(lane, slot);
        int col = g * H_ + nt * 16 + (lane & 15);
        float v = (k < H_) ? W[(long)k * FH_ + col] : U[(long)(k - H_) * FH_ + col];
        wcat[j] = f2b(v);
    }
    // Wc = W + U fragments, layout [nt][g][kt32][512]
    for (long j = tid; j < 64L * 4 * 32 * 512; j += nth) {
        int e = (int)(j & 511); int lane = e >> 4; int slot = e & 15;
        long t = j >> 9; int kt = (int)(t & 31); t >>= 5; int g = (int)(t & 3); int nt = (int)(t >> 2);
        int k   = kt * 32 + bK(lane, slot);
        int col = g * H_ + nt * 16 + (lane & 15);
        wc[j] = f2b(W[(long)k * FH_ + col] + U[(long)k * FH_ + col]);
    }
    // Wd fragments (emission), layout [nt][kt][512]
    for (long j = tid; j < 32L * 32 * 512; j += nth) {
        int e = (int)(j & 511); int lane = e >> 4; int slot = e & 15;
        long t = j >> 9; int kt = (int)(t & 31); int nt = (int)(t >> 5);
        int k   = kt * 32 + bK(lane, slot);
        int col = nt * 16 + (lane & 15);
        wdf[j] = f2b(Wd[(long)k * F_ + col]);
    }
    // [x0, h0] bf16, row-major [B, 2H]
    for (long j = tid; j < (long)B_ * 2 * H_; j += nth) {
        int col = (int)(j & (2 * H_ - 1)); long r = j >> 11;
        float v = (col < H_) ? x0[r * H_ + col] : h0[r * H_ + (col - H_)];
        xh[j] = f2b(v);
    }
    // grid-barrier counters (must be re-zeroed every launch)
    for (long j = tid; j < T_; j += nth) bar[j] = 0u;
}

// ---------------------------------------------------------------------------
// Grid-wide barrier (all 128 persistent blocks are co-resident).
// ---------------------------------------------------------------------------
static __device__ __forceinline__ void grid_barrier(unsigned* bar, int slot) {
    __syncthreads();
    if (threadIdx.x == 0) {
        __threadfence();  // release: h stores visible at device scope
        __hip_atomic_fetch_add(&bar[slot], 1u, __ATOMIC_ACQ_REL, __HIP_MEMORY_SCOPE_AGENT);
        while (__hip_atomic_load(&bar[slot], __ATOMIC_ACQUIRE, __HIP_MEMORY_SCOPE_AGENT) < NBLOCKS) {
            __builtin_amdgcn_s_sleep(1);
        }
        __threadfence();  // acquire for the whole WGP's caches
    }
    __syncthreads();
}

// ---------------------------------------------------------------------------
// Persistent recurrent kernel. Grid (2, 64), 256 thr = 8 waves.
// Wave tile: 16 rows x 16 cols x 4 gates. Block: M=128, N=16 (x4 gates).
// ---------------------------------------------------------------------------
__global__ void __launch_bounds__(256)
lstm_persistent(const __bf16* __restrict__ xh,    // [B, 2H] bf16
                const __bf16* __restrict__ wcat,  // step-0 fragments
                const __bf16* __restrict__ wc,    // steady-state fragments
                const float*  __restrict__ bias,  // [4096]
                const float*  __restrict__ c0,    // [B, H]
                __bf16*       __restrict__ seq,   // [B, T, H] bf16
                unsigned*     __restrict__ bar)   // [T]
{
    extern __shared__ __bf16 ldsW[];              // 4*32*512 bf16 = 128 KB

    const int lane  = threadIdx.x & 31;
    const int w     = threadIdx.x >> 5;
    const int mt    = blockIdx.x * 8 + w;         // 0..15
    const int nt    = blockIdx.y;                 // 0..63
    const int mbase = mt * 16;
    const int nlane = lane & 15;
    const bool hi   = lane >= 16;
    const int col   = nt * 16 + nlane;

    // Bias hoisted across all timesteps.
    const float bi = bias[0 * H_ + col];
    const float bf = bias[1 * H_ + col];
    const float bg = bias[2 * H_ + col];
    const float bo = bias[3 * H_ + col];

    // ---- kick off the one-time weight-slice copy into LDS (128 KB) --------
    {
        const __bf16* src = wc + (size_t)nt * (4 * 32 * 512);
#ifdef USE_ASYNC_LDS
#pragma unroll
        for (int i = 0; i < 32; ++i) {            // 256 thr * 32 * 16B = 128 KB
            const int e = (i * 256 + (int)threadIdx.x) * 8;
            __builtin_amdgcn_global_load_async_to_lds_b128(
                (gv4i_t*)(src + e), (lv4i_t*)(ldsW + e), 0, 0);
        }
#else
        for (int i = 0; i < 32; ++i) {
            const int e = (i * 256 + (int)threadIdx.x) * 8;
            *(v8bf*)&ldsW[e] = *(const v8bf*)(src + e);
        }
#endif
    }

    float creg[8];

    // ---- step 0: stream [W;U] fragments from global (K = 2048) ------------
    {
        v8f ai = {}; v8f af = {}; v8f ag = {}; v8f ao = {};
        const __bf16* arow = xh + (long)(mbase + nlane) * (2 * H_);
        const __bf16* b0 = wcat + ((long)(0 * 64 + nt) * 64) * 512;
        const __bf16* b1 = wcat + ((long)(1 * 64 + nt) * 64) * 512;
        const __bf16* b2 = wcat + ((long)(2 * 64 + nt) * 64) * 512;
        const __bf16* b3 = wcat + ((long)(3 * 64 + nt) * 64) * 512;
        for (int kt = 0; kt < 64; ++kt) {
            v16bf a = load_a_frag(arow, kt * 32, hi);
            if (kt + 2 < 64)
                __builtin_prefetch(b0 + (long)(kt + 2) * 512 + lane * 16, 0, 1);
            v16bf fi = load_b_frag(b0 + (long)kt * 512, lane);
            v16bf ff = load_b_frag(b1 + (long)kt * 512, lane);
            v16bf fg = load_b_frag(b2 + (long)kt * 512, lane);
            v16bf fo = load_b_frag(b3 + (long)kt * 512, lane);
            ai = __builtin_amdgcn_wmma_f32_16x16x32_bf16(false, a, false, fi, (short)0, ai, false, false);
            af = __builtin_amdgcn_wmma_f32_16x16x32_bf16(false, a, false, ff, (short)0, af, false, false);
            ag = __builtin_amdgcn_wmma_f32_16x16x32_bf16(false, a, false, fg, (short)0, ag, false, false);
            ao = __builtin_amdgcn_wmma_f32_16x16x32_bf16(false, a, false, fo, (short)0, ao, false, false);
        }
#pragma unroll
        for (int r = 0; r < 8; ++r) {
            const int row = mbase + (hi ? r + 8 : r);
            float zi = ai[r] + bi, zf = af[r] + bf, zg = ag[r] + bg, zo = ao[r] + bo;
            float cn = sigm(zf) * c0[(long)row * H_ + col] + sigm(zi) * tanh_fast(zg);
            float hn = sigm(zo) * tanh_fast(cn);
            creg[r] = cn;
            seq[(long)row * SEQ_STRIDE + 0 * H_ + col] = f2b(hn);
        }
    }

    // Weight slice must be in LDS before the steady-state loop.
#ifdef USE_ASYNC_LDS
    __builtin_amdgcn_s_wait_asynccnt(0);
#endif
    grid_barrier(bar, 0);

    // ---- steps 1..T-1: weights from LDS, h from seq ------------------------
    for (int t = 1; t < T_; ++t) {
        v8f ai = {}; v8f af = {}; v8f ag = {}; v8f ao = {};
        const __bf16* arow = seq + (long)(t - 1) * H_ + (long)(mbase + nlane) * SEQ_STRIDE;
        for (int kt = 0; kt < 32; ++kt) {
            v16bf a = load_a_frag(arow, kt * 32, hi);
            v16bf fi = load_b_frag(&ldsW[(0 * 32 + kt) * 512], lane);
            v16bf ff = load_b_frag(&ldsW[(1 * 32 + kt) * 512], lane);
            v16bf fg = load_b_frag(&ldsW[(2 * 32 + kt) * 512], lane);
            v16bf fo = load_b_frag(&ldsW[(3 * 32 + kt) * 512], lane);
            ai = __builtin_amdgcn_wmma_f32_16x16x32_bf16(false, a, false, fi, (short)0, ai, false, false);
            af = __builtin_amdgcn_wmma_f32_16x16x32_bf16(false, a, false, ff, (short)0, af, false, false);
            ag = __builtin_amdgcn_wmma_f32_16x16x32_bf16(false, a, false, fg, (short)0, ag, false, false);
            ao = __builtin_amdgcn_wmma_f32_16x16x32_bf16(false, a, false, fo, (short)0, ao, false, false);
        }
#pragma unroll
        for (int r = 0; r < 8; ++r) {
            const int row = mbase + (hi ? r + 8 : r);
            float zi = ai[r] + bi, zf = af[r] + bf, zg = ag[r] + bg, zo = ao[r] + bo;
            float cn = sigm(zf) * creg[r] + sigm(zi) * tanh_fast(zg);
            float hn = sigm(zo) * tanh_fast(cn);
            creg[r] = cn;
            seq[(long)row * SEQ_STRIDE + (long)t * H_ + col] = f2b(hn);
        }
        grid_barrier(bar, t);
    }
}

// ---------------------------------------------------------------------------
// Emission GEMM: out[B*T, F] = seq[B*T, H](bf16) @ Wd(bf16) + bd, f32 out.
// Grid (1024, 8), 8 waves/block, wave tile 16x16, block tile 32x64.
// ---------------------------------------------------------------------------
__global__ void __launch_bounds__(256)
emit_gemm_kernel(const __bf16* __restrict__ A,    // [32768, 1024]
                 const __bf16* __restrict__ Bf,   // Wd fragments
                 const float*  __restrict__ bd,   // [512]
                 float*        __restrict__ out)  // [32768, 512]
{
    const int lane  = threadIdx.x & 31;
    const int w     = threadIdx.x >> 5;
    const int mt    = blockIdx.x * 2 + (w >> 2);
    const int nt    = blockIdx.y * 4 + (w & 3);
    const int mbase = mt * 16;
    const int nlane = lane & 15;
    const bool hi   = lane >= 16;

    v8f acc = {};
    const __bf16* arow = A + (long)(mbase + nlane) * H_;
    const __bf16* bcol = Bf + ((long)nt * 32) * 512;

    for (int kt = 0; kt < 32; ++kt) {
        v16bf a = load_a_frag(arow, kt * 32, hi);
        if (kt + 2 < 32)
            __builtin_prefetch(bcol + (long)(kt + 2) * 512 + lane * 16, 0, 1);
        v16bf bfr = load_b_frag(bcol + (long)kt * 512, lane);
        acc = __builtin_amdgcn_wmma_f32_16x16x32_bf16(false, a, false, bfr, (short)0, acc, false, false);
    }

    const int col = nt * 16 + nlane;
    const float bias = bd[col];
#pragma unroll
    for (int r = 0; r < 8; ++r) {
        const int row = mbase + (hi ? r + 8 : r);
        out[(long)row * F_ + col] = acc[r] + bias;
    }
}

// ---------------------------------------------------------------------------
extern "C" void kernel_launch(void* const* d_in, const int* in_sizes, int n_in,
                              void* d_out, int out_size, void* d_ws, size_t ws_size,
                              hipStream_t stream) {
    (void)in_sizes; (void)n_in; (void)out_size; (void)ws_size;

    const float* x0 = (const float*)d_in[0];
    const float* h0 = (const float*)d_in[1];
    const float* c0 = (const float*)d_in[2];
    const float* W  = (const float*)d_in[3];
    const float* U  = (const float*)d_in[4];
    const float* bb = (const float*)d_in[5];
    const float* Wd = (const float*)d_in[6];
    const float* bd = (const float*)d_in[7];

    char* ws = (char*)d_ws;
    size_t off = 0;
    auto carve = [&](size_t bytes) -> char* {
        char* p = ws + off;
        off = (off + bytes + 255) & ~(size_t)255;
        return p;
    };
    __bf16*   seq  = (__bf16*)  carve((size_t)B_ * T_ * H_ * 2);        // 64 MB
    __bf16*   wcat = (__bf16*)  carve((size_t)4 * 64 * 64 * 512 * 2);   // 16 MB
    __bf16*   wc   = (__bf16*)  carve((size_t)64 * 4 * 32 * 512 * 2);   //  8 MB
    __bf16*   wdf  = (__bf16*)  carve((size_t)32 * 32 * 512 * 2);       //  1 MB
    __bf16*   xh   = (__bf16*)  carve((size_t)B_ * 2 * H_ * 2);         //  1 MB
    unsigned* bar  = (unsigned*)carve((size_t)T_ * 4);

    prep_kernel<<<2048, 256, 0, stream>>>(x0, h0, W, U, Wd, wcat, wc, wdf, xh, bar);

    // 128 co-resident blocks; 128 KB dynamic LDS each (WGP has 320 KB).
    lstm_persistent<<<dim3(2, 64), 256, 4 * 32 * 512 * sizeof(__bf16), stream>>>(
        xh, wcat, wc, bb, c0, seq, bar);

    emit_gemm_kernel<<<dim3((B_ * T_) / 32, F_ / 64), 256, 0, stream>>>(
        seq, wdf, bd, (float*)d_out);
}
